// STARVisionDetector_75952201662652
// MI455X (gfx1250) — matrix operand order repaired and checked by reading, not compiled, not run on hardware
//
#include <hip/hip_runtime.h>
#include <math.h>

// ---------------------------------------------------------------------------
// Types
// ---------------------------------------------------------------------------
typedef __attribute__((ext_vector_type(16))) __bf16 v16bf;
typedef __attribute__((ext_vector_type(8)))  float  v8f;
typedef __attribute__((ext_vector_type(4)))  unsigned int u32x4;

#define DIM      768
#define GRID_N   14
#define P_PATCH  196
#define BATCH    64
#define NTOK     (BATCH * P_PATCH)   // 12544
#define NBOX     32
#define NCLS     80

#define WMMA_BF16(a, b, c) \
  __builtin_amdgcn_wmma_f32_16x16x32_bf16(false, (a), false, (b), (short)0, (c), false, false)

__device__ __forceinline__ unsigned short f2bf(float f) {
  union { __bf16 h; unsigned short u; } cv;
  cv.h = (__bf16)f;
  return cv.u;
}

// ---------------------------------------------------------------------------
// Weight prep
// ---------------------------------------------------------------------------
__global__ void k_convert_bf16(const float* __restrict__ src,
                               unsigned short* __restrict__ dst, int n) {
  int i = blockIdx.x * blockDim.x + threadIdx.x;
  if (i < n) dst[i] = f2bf(src[i]);
}

// dst[rows_out, DIM] = transpose(src[DIM, ncols]), zero padded past ncols
__global__ void k_transpose_pad(const float* __restrict__ src,
                                unsigned short* __restrict__ dst,
                                int ncols, int rows_out) {
  int i = blockIdx.x * blockDim.x + threadIdx.x;
  int total = rows_out * DIM;
  if (i >= total) return;
  int n = i / DIM, k = i % DIM;
  dst[i] = (n < ncols) ? f2bf(src[k * ncols + n]) : (unsigned short)0;
}

// ---------------------------------------------------------------------------
// im2col: Xp[n=12544, k=768] bf16, k = c*256 + i*16 + j (matches patch_w)
// ---------------------------------------------------------------------------
__global__ void k_im2col(const float* __restrict__ imgs,
                         unsigned short* __restrict__ Xp) {
  int i = blockIdx.x * blockDim.x + threadIdx.x;
  if (i >= NTOK * DIM) return;
  int n = i / DIM, k = i % DIM;
  int b = n / P_PATCH, p = n % P_PATCH;
  int py = p / GRID_N, px = p % GRID_N;
  int c = k >> 8, r = (k >> 4) & 15, q = k & 15;
  Xp[i] = f2bf(imgs[((b * 3 + c) * 224 + py * 16 + r) * 224 + px * 16 + q]);
}

// ---------------------------------------------------------------------------
// WMMA fragment loaders (layouts per CDNA5 ISA 7.12.2, wave32)
// A 16x32 bf16: lane l -> row M=l%16, half=l/16; e0..7 -> K=k+half*8+e,
//               e8..15 -> K=k+16+half*8+(e-8)  (two 16B loads)
// B 32x16 bf16 from N-major Bt[n,k]: lane l -> col N=l%16, same K order
//               as A: e0..7 -> K=k+half*8+e, e8..15 -> K=k+16+half*8+(e-8)
// ---------------------------------------------------------------------------
__device__ __forceinline__ v16bf frag_a(const unsigned short* __restrict__ row,
                                        int k, int hf) {
  union { u32x4 u[2]; v16bf v; } f;
  f.u[0] = *(const u32x4*)(row + k + hf * 8);
  f.u[1] = *(const u32x4*)(row + k + 16 + hf * 8);
  return f.v;
}
__device__ __forceinline__ v16bf frag_b(const unsigned short* __restrict__ row,
                                        int k, int hf) {
  union { u32x4 u[2]; v16bf v; } f;
  f.u[0] = *(const u32x4*)(row + k + hf * 8);
  f.u[1] = *(const u32x4*)(row + k + 16 + hf * 8);
  return f.v;
}

// ---------------------------------------------------------------------------
// Core 32(M) x 64(N) x 768(K) wave tile, software-pipelined:
// "next" fragments are loaded before the current WMMAs issue, so the
// s_wait before each wmma only covers last-iteration loads (12 in flight).
// 8 v_wmma per 12 loads. EXEC uniform throughout.
// ---------------------------------------------------------------------------
__device__ __forceinline__ void gemm32x64(
    const unsigned short* __restrict__ A, const unsigned short* __restrict__ Bt,
    int m0, int n0, int hf, int lr, v8f acc[2][4]) {
  const unsigned short* Ar0 = A + (size_t)(m0 + lr) * DIM;
  const unsigned short* Ar1 = A + (size_t)(m0 + 16 + lr) * DIM;
  const unsigned short* Br  = Bt + (size_t)(n0 + lr) * DIM;

  v16bf a0 = frag_a(Ar0, 0, hf);
  v16bf a1 = frag_a(Ar1, 0, hf);
  v16bf b0 = frag_b(Br,            0, hf);
  v16bf b1 = frag_b(Br + 16 * DIM, 0, hf);
  v16bf b2 = frag_b(Br + 32 * DIM, 0, hf);
  v16bf b3 = frag_b(Br + 48 * DIM, 0, hf);

#pragma unroll 4
  for (int k = 0; k < DIM; k += 32) {
    int kn = (k + 32 < DIM) ? k + 32 : 0;  // harmless redundant load on last iter
    v16bf na0 = frag_a(Ar0, kn, hf);
    v16bf na1 = frag_a(Ar1, kn, hf);
    v16bf nb0 = frag_b(Br,            kn, hf);
    v16bf nb1 = frag_b(Br + 16 * DIM, kn, hf);
    v16bf nb2 = frag_b(Br + 32 * DIM, kn, hf);
    v16bf nb3 = frag_b(Br + 48 * DIM, kn, hf);

    acc[0][0] = WMMA_BF16(a0, b0, acc[0][0]);
    acc[0][1] = WMMA_BF16(a0, b1, acc[0][1]);
    acc[0][2] = WMMA_BF16(a0, b2, acc[0][2]);
    acc[0][3] = WMMA_BF16(a0, b3, acc[0][3]);
    acc[1][0] = WMMA_BF16(a1, b0, acc[1][0]);
    acc[1][1] = WMMA_BF16(a1, b1, acc[1][1]);
    acc[1][2] = WMMA_BF16(a1, b2, acc[1][2]);
    acc[1][3] = WMMA_BF16(a1, b3, acc[1][3]);

    a0 = na0; a1 = na1;
    b0 = nb0; b1 = nb1; b2 = nb2; b3 = nb3;
  }
}

// ---------------------------------------------------------------------------
// Patch-embed GEMM: E[12544,768] = Xp x Wp^T + patch_b + pos_embed (fp32 out)
// block = 128 threads = 4 waves, wave -> 32x64 tile; grid (98, 12)
// ---------------------------------------------------------------------------
__global__ __launch_bounds__(128) void k_gemm_embed(
    const unsigned short* __restrict__ A, const unsigned short* __restrict__ Bt,
    const float* __restrict__ bias, const float* __restrict__ pos,
    float* __restrict__ Ef) {
  int wave = threadIdx.x >> 5, lane = threadIdx.x & 31;
  int hf = lane >> 4, lr = lane & 15;
  int m0 = blockIdx.x * 128 + wave * 32;
  int n0 = blockIdx.y * 64;
  v8f acc[2][4] = {};
  gemm32x64(A, Bt, m0, n0, hf, lr, acc);
#pragma unroll
  for (int mt = 0; mt < 2; mt++) {
#pragma unroll
    for (int r = 0; r < 8; r++) {
      int m = m0 + mt * 16 + r + hf * 8;
      const float* posr = pos + (size_t)(m % P_PATCH) * DIM;
      float* outr = Ef + (size_t)m * DIM;
#pragma unroll
      for (int nt = 0; nt < 4; nt++) {
        int n = n0 + nt * 16 + lr;
        outr[n] = acc[mt][nt][r] + bias[n] + posr[n];
      }
    }
  }
}

// ---------------------------------------------------------------------------
// Fused 3x hidden GEMM: H_z = bf16(silu(Xn x W1t_z^T + b1_z)), z = cls/obj/bbox
// grid (98, 12, 3); z-select is uniform (SGPR), EXEC stays all-ones.
// ---------------------------------------------------------------------------
__global__ __launch_bounds__(128) void k_gemm_h3(
    const unsigned short* __restrict__ Xn,
    const unsigned short* __restrict__ B0, const float* __restrict__ bias0,
    unsigned short* __restrict__ H0,
    const unsigned short* __restrict__ B1, const float* __restrict__ bias1,
    unsigned short* __restrict__ H1,
    const unsigned short* __restrict__ B2, const float* __restrict__ bias2,
    unsigned short* __restrict__ H2) {
  int z = blockIdx.z;
  const unsigned short* Bt = (z == 0) ? B0 : (z == 1) ? B1 : B2;
  const float* bias        = (z == 0) ? bias0 : (z == 1) ? bias1 : bias2;
  unsigned short* H        = (z == 0) ? H0 : (z == 1) ? H1 : H2;

  int wave = threadIdx.x >> 5, lane = threadIdx.x & 31;
  int hf = lane >> 4, lr = lane & 15;
  int m0 = blockIdx.x * 128 + wave * 32;
  int n0 = blockIdx.y * 64;
  v8f acc[2][4] = {};
  gemm32x64(Xn, Bt, m0, n0, hf, lr, acc);
#pragma unroll
  for (int mt = 0; mt < 2; mt++) {
#pragma unroll
    for (int r = 0; r < 8; r++) {
      int m = m0 + mt * 16 + r + hf * 8;
      unsigned short* outr = H + (size_t)m * DIM;
#pragma unroll
      for (int nt = 0; nt < 4; nt++) {
        int n = n0 + nt * 16 + lr;
        float v = acc[mt][nt][r] + bias[n];
        outr[n] = f2bf(v / (1.f + __expf(-v)));  // silu
      }
    }
  }
}

// ---------------------------------------------------------------------------
// Fused 3x head GEMM; wave -> 16x16 tile; grid (196, 5, 3).
// z=0: cls  (80 cols, 5 n-tiles, logits)
// z=1: obj  (1 col,  padded Bt, logits)
// z=2: bbox (4 cols, padded Bt, sigmoid)
// ---------------------------------------------------------------------------
__global__ __launch_bounds__(128) void k_gemm_head3(
    const unsigned short* __restrict__ Hc, const unsigned short* __restrict__ Ho,
    const unsigned short* __restrict__ Hb,
    const unsigned short* __restrict__ Wc, const unsigned short* __restrict__ Wo,
    const unsigned short* __restrict__ Wb,
    const float* __restrict__ bc, const float* __restrict__ bo,
    const float* __restrict__ bbv,
    float* __restrict__ oc, float* __restrict__ oo, float* __restrict__ ob) {
  int z = blockIdx.z;
  if (z != 0 && blockIdx.y != 0) return;  // uniform exit: obj/bbox have 1 n-tile
  const unsigned short* A  = (z == 0) ? Hc : (z == 1) ? Ho : Hb;
  const unsigned short* Bt = (z == 0) ? Wc : (z == 1) ? Wo : Wb;
  const float* bias        = (z == 0) ? bc : (z == 1) ? bo : bbv;
  float* out               = (z == 0) ? oc : (z == 1) ? oo : ob;
  int ncols                = (z == 0) ? NCLS : (z == 1) ? 1 : 4;
  int act                  = (z == 2) ? 1 : 0;

  int wave = threadIdx.x >> 5, lane = threadIdx.x & 31;
  int hf = lane >> 4, lr = lane & 15;
  int m0 = blockIdx.x * 64 + wave * 16;
  int n0 = blockIdx.y * 16;
  const unsigned short* Ar = A + (size_t)(m0 + lr) * DIM;
  const unsigned short* Br = Bt + (size_t)(n0 + lr) * DIM;

  v8f acc = {};
  v16bf a = frag_a(Ar, 0, hf);
  v16bf b = frag_b(Br, 0, hf);
#pragma unroll 4
  for (int k = 0; k < DIM; k += 32) {
    int kn = (k + 32 < DIM) ? k + 32 : 0;
    v16bf na = frag_a(Ar, kn, hf);
    v16bf nb = frag_b(Br, kn, hf);
    acc = WMMA_BF16(a, b, acc);
    a = na; b = nb;
  }
  int n = n0 + lr;
  if (n < ncols) {
#pragma unroll
    for (int r = 0; r < 8; r++) {
      int m = m0 + r + hf * 8;
      float v = acc[r] + bias[n];
      if (act) v = 1.f / (1.f + __expf(-v));
      out[(size_t)m * ncols + n] = v;
    }
  }
}

// ---------------------------------------------------------------------------
// RMSNorm: one wave per row; Xn = bf16( E * rsqrt(mean(E^2)+1e-6) * w )
// ---------------------------------------------------------------------------
__global__ __launch_bounds__(256) void k_rmsnorm(
    const float* __restrict__ E, const float* __restrict__ w,
    unsigned short* __restrict__ Xn) {
  int wave = threadIdx.x >> 5, lane = threadIdx.x & 31;
  int row = blockIdx.x * 8 + wave;
  const float* e = E + (size_t)row * DIM;
  float ss = 0.f;
  for (int i = lane; i < DIM; i += 32) { float v = e[i]; ss += v * v; }
#pragma unroll
  for (int o = 16; o > 0; o >>= 1) ss += __shfl_xor(ss, o, 32);
  float sc = rsqrtf(ss * (1.f / DIM) + 1e-6f);
  for (int i = lane; i < DIM; i += 32)
    Xn[(size_t)row * DIM + i] = f2bf(e[i] * sc * w[i]);
}

// ---------------------------------------------------------------------------
// Loss
// ---------------------------------------------------------------------------
__global__ void k_zero(float* p) {
  if (threadIdx.x == 0 && blockIdx.x == 0) *p = 0.f;
}

__device__ float blk_sum(float v, float* red) {
  int t = threadIdx.x;
  red[t] = v;
  __syncthreads();
  for (int s = 128; s > 0; s >>= 1) {
    if (t < s) red[t] += red[t + s];
    __syncthreads();
  }
  float r = red[0];
  __syncthreads();
  return r;
}

__global__ __launch_bounds__(256) void k_loss(
    const float* __restrict__ cls, const float* __restrict__ obj,
    const float* __restrict__ bbox, const float* __restrict__ boxes,
    const int* __restrict__ labels, float* __restrict__ loss_out) {
  __shared__ int last[P_PATCH];
  __shared__ float red[256];
  int b = blockIdx.x, t = threadIdx.x;
  if (t < P_PATCH) last[t] = -1;
  __syncthreads();
  if (t < NBOX) {  // last-write-wins == max box index per patch
    const float* bx = boxes + (size_t)(b * NBOX + t) * 4;
    int gx = (int)(bx[0] * GRID_N); gx = gx < 0 ? 0 : (gx > GRID_N - 1 ? GRID_N - 1 : gx);
    int gy = (int)(bx[1] * GRID_N); gy = gy < 0 ? 0 : (gy > GRID_N - 1 ? GRID_N - 1 : gy);
    atomicMax(&last[gy * GRID_N + gx], t);
  }
  __syncthreads();
  float s_vf = 0.f, s_l1 = 0.f, s_giou = 0.f, s_nll = 0.f, s_foc = 0.f;
  if (t < P_PATCH) {
    int lw = last[t];
    float vf = lw >= 0 ? 1.f : 0.f;
    int safe = lw < 0 ? 0 : lw;
    const float* tb = boxes + (size_t)(b * NBOX + safe) * 4;
    int lab = labels[b * NBOX + safe];
    const float* pb = bbox + (size_t)(b * P_PATCH + t) * 4;
    float l1 = 0.25f * (fabsf(pb[0] - tb[0]) + fabsf(pb[1] - tb[1]) +
                        fabsf(pb[2] - tb[2]) + fabsf(pb[3] - tb[3]));
    float p0 = pb[0] - pb[2] * 0.5f, p1 = pb[1] - pb[3] * 0.5f;
    float p2 = pb[0] + pb[2] * 0.5f, p3 = pb[1] + pb[3] * 0.5f;
    float t0 = tb[0] - tb[2] * 0.5f, t1 = tb[1] - tb[3] * 0.5f;
    float t2 = tb[0] + tb[2] * 0.5f, t3 = tb[1] + tb[3] * 0.5f;
    float ix1 = fmaxf(p0, t0), iy1 = fmaxf(p1, t1);
    float ix2 = fminf(p2, t2), iy2 = fminf(p3, t3);
    float inter = fmaxf(ix2 - ix1, 0.f) * fmaxf(iy2 - iy1, 0.f);
    float ap = fmaxf(p2 - p0, 0.f) * fmaxf(p3 - p1, 0.f);
    float at = fmaxf(t2 - t0, 0.f) * fmaxf(t3 - t1, 0.f);
    float uni = ap + at - inter;
    float iou = inter / (uni + 1e-7f);
    float ex1 = fminf(p0, t0), ey1 = fminf(p1, t1);
    float ex2 = fmaxf(p2, t2), ey2 = fmaxf(p3, t3);
    float enc = fmaxf(ex2 - ex1, 0.f) * fmaxf(ey2 - ey1, 0.f);
    float giou = iou - (enc - uni) / (enc + 1e-7f);
    const float* cl = cls + (size_t)(b * P_PATCH + t) * NCLS;
    float mx = cl[0];
    for (int j = 1; j < NCLS; j++) mx = fmaxf(mx, cl[j]);
    float se = 0.f;
    for (int j = 0; j < NCLS; j++) se += __expf(cl[j] - mx);
    float nll = -(cl[lab] - mx - __logf(se));
    float lg = obj[b * P_PATCH + t];
    float ce = fmaxf(lg, 0.f) - lg * vf + log1pf(__expf(-fabsf(lg)));
    float pr = 1.f / (1.f + __expf(-lg));
    float pt = pr * vf + (1.f - pr) * (1.f - vf);
    float af = 0.25f * vf + 0.75f * (1.f - vf);
    float om = 1.f - pt;
    s_foc = af * om * om * ce;
    s_vf = vf; s_l1 = l1 * vf; s_giou = giou * vf; s_nll = nll * vf;
  }
  float vfs = blk_sum(s_vf, red);
  float l1s = blk_sum(s_l1, red);
  float gis = blk_sum(s_giou, red);
  float nls = blk_sum(s_nll, red);
  float fcs = blk_sum(s_foc, red);
  if (t == 0) {
    float n = fmaxf(vfs, 1.f);
    float box_term = (vfs > 0.f)
        ? (nls / n + 5.f * (l1s / n) + 2.f * (1.f - gis / n)) : 0.f;
    float loss = box_term + fcs / (float)P_PATCH;
    atomicAdd(loss_out, loss * (1.f / BATCH));
  }
}

// ---------------------------------------------------------------------------
// Host launcher
// ---------------------------------------------------------------------------
extern "C" void kernel_launch(void* const* d_in, const int* in_sizes, int n_in,
                              void* d_out, int out_size, void* d_ws, size_t ws_size,
                              hipStream_t stream) {
  (void)in_sizes; (void)n_in; (void)out_size; (void)ws_size;
  const float* imgs    = (const float*)d_in[0];
  const float* boxes   = (const float*)d_in[1];
  const int*   labels  = (const int*)d_in[2];
  const float* patch_w = (const float*)d_in[3];
  const float* patch_b = (const float*)d_in[4];
  const float* pos     = (const float*)d_in[5];
  const float* norm_w  = (const float*)d_in[6];
  const float* cls_w1  = (const float*)d_in[7];
  const float* cls_b1  = (const float*)d_in[8];
  const float* cls_w2  = (const float*)d_in[9];
  const float* cls_b2  = (const float*)d_in[10];
  const float* obj_w1  = (const float*)d_in[11];
  const float* obj_b1  = (const float*)d_in[12];
  const float* obj_w2  = (const float*)d_in[13];
  const float* obj_b2  = (const float*)d_in[14];
  const float* bbox_w1 = (const float*)d_in[15];
  const float* bbox_b1 = (const float*)d_in[16];
  const float* bbox_w2 = (const float*)d_in[17];
  const float* bbox_b2 = (const float*)d_in[18];

  float* out      = (float*)d_out;
  float* out_cls  = out;
  float* out_obj  = out + (size_t)BATCH * P_PATCH * NCLS;     // 1003520
  float* out_bbox = out_obj + (size_t)BATCH * P_PATCH;        // 1016064
  float* out_loss = out_bbox + (size_t)BATCH * P_PATCH * 4;   // 1066240

  // workspace layout (256B-aligned slabs), ~82 MB total
  char* wsp = (char*)d_ws;
  size_t off = 0;
  auto nxt = [&](size_t bytes) {
    size_t r = off;
    off = (off + bytes + 255) & ~(size_t)255;
    return r;
  };
  unsigned short* Xp   = (unsigned short*)(wsp + nxt((size_t)NTOK * DIM * 2));
  unsigned short* Wp   = (unsigned short*)(wsp + nxt((size_t)DIM * DIM * 2));
  unsigned short* Wc1t = (unsigned short*)(wsp + nxt((size_t)DIM * DIM * 2));
  unsigned short* Wo1t = (unsigned short*)(wsp + nxt((size_t)DIM * DIM * 2));
  unsigned short* Wb1t = (unsigned short*)(wsp + nxt((size_t)DIM * DIM * 2));
  unsigned short* Wc2t = (unsigned short*)(wsp + nxt((size_t)NCLS * DIM * 2));
  unsigned short* Wo2t = (unsigned short*)(wsp + nxt((size_t)16 * DIM * 2));
  unsigned short* Wb2t = (unsigned short*)(wsp + nxt((size_t)16 * DIM * 2));
  float*          E    = (float*)(wsp + nxt((size_t)NTOK * DIM * 4));
  unsigned short* Xn   = (unsigned short*)(wsp + nxt((size_t)NTOK * DIM * 2));
  // dead-slab aliases for the three hidden activations (bf16, 19.3 MB each):
  unsigned short* Hc = Xp;                                   // Xp dead after embed GEMM
  unsigned short* Ho = (unsigned short*)E;                   // E dead after rmsnorm
  unsigned short* Hb = (unsigned short*)E + (size_t)NTOK * DIM;

  const int THR = 256;
  dim3 gemm_grid(NTOK / 128, DIM / 64);      // (98, 12)

  // weight prep
  k_convert_bf16<<<(DIM * DIM + THR - 1) / THR, THR, 0, stream>>>(patch_w, Wp, DIM * DIM);
  k_transpose_pad<<<(DIM * DIM + THR - 1) / THR, THR, 0, stream>>>(cls_w1, Wc1t, DIM, DIM);
  k_transpose_pad<<<(DIM * DIM + THR - 1) / THR, THR, 0, stream>>>(obj_w1, Wo1t, DIM, DIM);
  k_transpose_pad<<<(DIM * DIM + THR - 1) / THR, THR, 0, stream>>>(bbox_w1, Wb1t, DIM, DIM);
  k_transpose_pad<<<(NCLS * DIM + THR - 1) / THR, THR, 0, stream>>>(cls_w2, Wc2t, NCLS, NCLS);
  k_transpose_pad<<<(16 * DIM + THR - 1) / THR, THR, 0, stream>>>(obj_w2, Wo2t, 1, 16);
  k_transpose_pad<<<(16 * DIM + THR - 1) / THR, THR, 0, stream>>>(bbox_w2, Wb2t, 4, 16);

  // patch embed (im2col GEMM) + pos, then rmsnorm
  k_im2col<<<((size_t)NTOK * DIM + THR - 1) / THR, THR, 0, stream>>>(imgs, Xp);
  k_gemm_embed<<<gemm_grid, 128, 0, stream>>>(Xp, Wp, patch_b, pos, E);
  k_rmsnorm<<<NTOK / 8, 256, 0, stream>>>(E, norm_w, Xn);

  // three hidden layers in one launch (z = cls/obj/bbox)
  k_gemm_h3<<<dim3(NTOK / 128, DIM / 64, 3), 128, 0, stream>>>(
      Xn, Wc1t, cls_b1, Hc, Wo1t, obj_b1, Ho, Wb1t, bbox_b1, Hb);

  // three heads in one launch
  k_gemm_head3<<<dim3(NTOK / 64, 5, 3), 128, 0, stream>>>(
      Hc, Ho, Hb, Wc2t, Wo2t, Wb2t, cls_b2, obj_b2, bbox_b2,
      out_cls, out_obj, out_bbox);

  // loss
  k_zero<<<1, 1, 0, stream>>>(out_loss);
  k_loss<<<BATCH, 256, 0, stream>>>(out_cls, out_obj, out_bbox, boxes, labels, out_loss);
}
